// GATEncoder_70798240907297
// MI455X (gfx1250) — matrix operand order repaired and checked by reading, not compiled, not run on hardware
//
#include <hip/hip_runtime.h>
#include <hip/hip_bf16.h>
#include <cfloat>

typedef __attribute__((ext_vector_type(16))) _Float16 v16h;
typedef __attribute__((ext_vector_type(8)))  _Float16 v8h;
typedef __attribute__((ext_vector_type(8)))  float    v8f;

#define NNODES 50000
#define NEDGES 800000

// ---------------------------------------------------------------- utilities

__device__ inline void atomicMaxF(float* addr, float val) {
    // monotonic float ordering via int/uint bit tricks; addr initialized to -FLT_MAX
    if (val >= 0.0f) atomicMax((int*)addr, __float_as_int(val));
    else             atomicMin((unsigned int*)addr, __float_as_uint(val));
}

__global__ __launch_bounds__(256) void fill_f32(float* p, float v, long n) {
    long i = (long)blockIdx.x * blockDim.x + threadIdx.x;
    if (i < n) p[i] = v;
}

__global__ __launch_bounds__(256) void f32_to_f16(const float* in, _Float16* out, long n) {
    long i = (long)blockIdx.x * blockDim.x + threadIdx.x;
    if (i < n) out[i] = (_Float16)in[i];
}

// ELU (alpha=1) then convert to f16 for the next layer's GEMM input
__global__ __launch_bounds__(256) void elu_f32_to_f16(const float* in, _Float16* out, long n) {
    long i = (long)blockIdx.x * blockDim.x + threadIdx.x;
    if (i < n) {
        float v = in[i];
        float o = v > 0.0f ? v : expm1f(v);
        out[i] = (_Float16)o;
    }
}

// W [K, Nout] fp32 -> Wt [Nout, K] f16 (so B fragments are K-contiguous)
__global__ __launch_bounds__(256) void wt_convert(const float* W, _Float16* Wt, int K, int Nout) {
    int i = blockIdx.x * blockDim.x + threadIdx.x;
    if (i >= K * Nout) return;
    int k = i / Nout, n = i % Nout;
    Wt[(size_t)n * K + k] = (_Float16)W[i];
}

// ---------------------------------------------------------------- WMMA GEMM
// C[M,Nout] = A16[M,K] * Wt16[Nout,K]^T + bias.  One wave -> one 16x16 C tile.
// A frag (16-bit A 16x32): lane%16 = row M, lane/16 selects K-half (kb = 0/8),
//   halves 0..7 -> K = k0+kb+i, halves 8..15 -> K = k0+16+kb+(i-8).
// B frag (16-bit B 32x16): lane%16 = col N, lane/16 selects K-half (kb2 = 0/8),
//   halves 0..7 -> K = k0+kb2+i, halves 8..15 -> K = k0+16+kb2+(i-8).
__global__ __launch_bounds__(256) void gemm_f16_wmma(
        const _Float16* __restrict__ A, const _Float16* __restrict__ Bt,
        const float* __restrict__ bias, float* __restrict__ C, int K, int Nout) {
    const int lane  = threadIdx.x & 31;
    const int wave  = threadIdx.x >> 5;
    const int tileM = blockIdx.x * 16;
    const int tileN = blockIdx.y * 128 + wave * 16;
    if (tileN >= Nout) return;

    const int l16  = lane & 15;
    const int kbA  = (lane >> 4) * 8;
    const int kbB  = (lane >> 4) * 8;
    const _Float16* arow = A  + (size_t)(tileM + l16) * K;
    const _Float16* brow = Bt + (size_t)(tileN + l16) * K;

    v8f c = {};
    for (int k0 = 0; k0 < K; k0 += 32) {
        v8h alo = *(const v8h*)(arow + k0 + kbA);
        v8h ahi = *(const v8h*)(arow + k0 + 16 + kbA);
        v16h a;
#pragma unroll
        for (int i = 0; i < 8; ++i) { a[i] = alo[i]; a[8 + i] = ahi[i]; }
        v8h blo = *(const v8h*)(brow + k0 + kbB);
        v8h bhi = *(const v8h*)(brow + k0 + 16 + kbB);
        v16h b;
#pragma unroll
        for (int i = 0; i < 8; ++i) { b[i] = blo[i]; b[8 + i] = bhi[i]; }
        c = __builtin_amdgcn_wmma_f32_16x16x32_f16(
                /*neg_a=*/false, a, /*neg_b=*/false, b,
                /*c_mod=*/(short)0, c, /*reuse_a=*/false, /*reuse_b=*/false);
    }

    // C/D layout: VGPR r holds M = r + 8*(lane/16), N = lane%16
    const int mb = (lane >> 4) * 8;
    const int col = tileN + l16;
    const float bv = bias[col];
#pragma unroll
    for (int r = 0; r < 8; ++r) {
        C[(size_t)(tileM + mb + r) * Nout + col] = c[r] + bv;
    }
}

// ---------------------------------------------------------------- edge phase
// score[e,h] = sum_d attn[h,d] * leaky_relu(h[src,h,d] + h[dst,h,d], 0.2)
// one wave per edge; lanes cover D, shuffle-reduce; atomic segment max.
__global__ __launch_bounds__(256) void edge_score(
        const float* __restrict__ h, const float* __restrict__ attn,
        const int* __restrict__ src, const int* __restrict__ dst,
        float* __restrict__ score, float* __restrict__ mmax,
        int H, int D, int E) {
    int gw   = (int)(((long)blockIdx.x * blockDim.x + threadIdx.x) >> 5);
    int lane = threadIdx.x & 31;
    if (gw >= E) return;
    int s = src[gw], d = dst[gw];
    int F = H * D;
    const float* hs = h + (size_t)s * F;
    const float* hd = h + (size_t)d * F;
    for (int head = 0; head < H; ++head) {
        float acc = 0.0f;
        for (int dd = lane; dd < D; dd += 32) {
            float v = hs[head * D + dd] + hd[head * D + dd];
            v = v > 0.0f ? v : 0.2f * v;
            acc += attn[head * D + dd] * v;
        }
        for (int off = 16; off > 0; off >>= 1)
            acc += __shfl_xor(acc, off, 32);
        if (lane == 0) {
            score[(size_t)gw * H + head] = acc;
            atomicMaxF(&mmax[(size_t)d * H + head], acc);
        }
    }
}

// ex = exp(score - m[dst]); denom[dst] += ex   (ex stored in-place over score)
__global__ __launch_bounds__(256) void edge_exp(
        const float* __restrict__ mmax, const int* __restrict__ dst,
        float* __restrict__ score, float* __restrict__ denom, int H, int E) {
    long idx = (long)blockIdx.x * blockDim.x + threadIdx.x;
    if (idx >= (long)E * H) return;
    int e = (int)(idx / H), hh = (int)(idx % H);
    int d = dst[e];
    float ex = expf(score[idx] - mmax[(size_t)d * H + hh]);
    score[idx] = ex;
    atomicAdd(&denom[(size_t)d * H + hh], ex);
}

// agg[dst,h,d] += (ex/denom[dst,h]) * h[src,h,d]   — one wave per edge
__global__ __launch_bounds__(256) void edge_aggregate(
        const float* __restrict__ h, const float* __restrict__ ex,
        const float* __restrict__ denom,
        const int* __restrict__ src, const int* __restrict__ dst,
        float* __restrict__ agg, int H, int D, int E) {
    int gw   = (int)(((long)blockIdx.x * blockDim.x + threadIdx.x) >> 5);
    int lane = threadIdx.x & 31;
    if (gw >= E) return;
    int s = src[gw], d = dst[gw];
    int F = H * D;
    for (int f = lane; f < F; f += 32) {
        int head = f / D;
        float alpha = ex[(size_t)gw * H + head] / denom[(size_t)d * H + head];
        atomicAdd(&agg[(size_t)d * F + f], alpha * h[(size_t)s * F + f]);
    }
}

// graph max-pool: out[o] = max_n agg[n, o]   (H3==1 so head-mean is identity)
__global__ __launch_bounds__(256) void max_pool(
        const float* __restrict__ agg, float* __restrict__ out, int nnodes, int C) {
    int o = blockIdx.x;
    float best = -FLT_MAX;
    for (int n = threadIdx.x; n < nnodes; n += blockDim.x)
        best = fmaxf(best, agg[(size_t)n * C + o]);
    __shared__ float sh[256];
    sh[threadIdx.x] = best;
    __syncthreads();
    for (int s = 128; s > 0; s >>= 1) {
        if (threadIdx.x < s) sh[threadIdx.x] = fmaxf(sh[threadIdx.x], sh[threadIdx.x + s]);
        __syncthreads();
    }
    if (threadIdx.x == 0) out[o] = sh[0];
}

// ---------------------------------------------------------------- host side

static void run_gat_layer(const float* W, const float* b, const float* attn,
                          const int* src, const int* dst,
                          _Float16* xin, _Float16* wt, float* h, float* agg,
                          float* score, float* mmax, float* denom,
                          int K, int H, int D, hipStream_t stream) {
    const int F = H * D;
    // weights -> transposed f16
    wt_convert<<<(K * F + 255) / 256, 256, 0, stream>>>(W, wt, K, F);
    // GEMM: h = xin @ W + b
    dim3 g(NNODES / 16, (F + 127) / 128);
    gemm_f16_wmma<<<g, 256, 0, stream>>>(xin, wt, b, h, K, F);
    // reset segment buffers
    fill_f32<<<((long)NNODES * F + 255) / 256, 256, 0, stream>>>(agg, 0.0f, (long)NNODES * F);
    fill_f32<<<((long)NNODES * H + 255) / 256, 256, 0, stream>>>(denom, 0.0f, (long)NNODES * H);
    fill_f32<<<((long)NNODES * H + 255) / 256, 256, 0, stream>>>(mmax, -FLT_MAX, (long)NNODES * H);
    // edge softmax + aggregation
    const int edgeBlocks = (NEDGES * 32 + 255) / 256;   // one wave per edge
    edge_score<<<edgeBlocks, 256, 0, stream>>>(h, attn, src, dst, score, mmax, H, D, NEDGES);
    edge_exp<<<((long)NEDGES * H + 255) / 256, 256, 0, stream>>>(mmax, dst, score, denom, H, NEDGES);
    edge_aggregate<<<edgeBlocks, 256, 0, stream>>>(h, score, denom, src, dst, agg, H, D, NEDGES);
}

extern "C" void kernel_launch(void* const* d_in, const int* in_sizes, int n_in,
                              void* d_out, int out_size, void* d_ws, size_t ws_size,
                              hipStream_t stream) {
    const float* x     = (const float*)d_in[0];
    const int*   src   = (const int*)  d_in[1];
    const int*   dst   = (const int*)  d_in[2];
    const float* W1    = (const float*)d_in[3];
    const float* b1    = (const float*)d_in[4];
    const float* attn1 = (const float*)d_in[5];
    const float* W2    = (const float*)d_in[6];
    const float* b2    = (const float*)d_in[7];
    const float* attn2 = (const float*)d_in[8];
    const float* W3    = (const float*)d_in[9];
    const float* b3    = (const float*)d_in[10];
    const float* attn3 = (const float*)d_in[11];
    float* out = (float*)d_out;

    // workspace carve-up (~157 MB): all offsets are 32B-aligned
    float* h     = (float*)d_ws;                       // [N,256] fp32 GEMM output
    float* agg   = h     + (size_t)NNODES * 256;       // [N,256] fp32 aggregation
    float* score = agg   + (size_t)NNODES * 256;       // [E,8]   fp32 score -> ex
    float* mmax  = score + (size_t)NEDGES * 8;         // [N,8]   segment max
    float* denom = mmax  + (size_t)NNODES * 8;         // [N,8]   segment sum
    _Float16* xin = (_Float16*)(denom + (size_t)NNODES * 8); // [N,256] f16 GEMM input
    _Float16* wt  = xin + (size_t)NNODES * 256;        // [256,256] f16 W^T

    // initial input: x [N,128] fp32 -> f16
    f32_to_f16<<<((long)NNODES * 128 + 255) / 256, 256, 0, stream>>>(x, xin, (long)NNODES * 128);

    // layer 1: K=128, H=8, D=32
    run_gat_layer(W1, b1, attn1, src, dst, xin, wt, h, agg, score, mmax, denom,
                  128, 8, 32, stream);
    elu_f32_to_f16<<<((long)NNODES * 256 + 255) / 256, 256, 0, stream>>>(agg, xin, (long)NNODES * 256);

    // layer 2: K=256, H=8, D=32
    run_gat_layer(W2, b2, attn2, src, dst, xin, wt, h, agg, score, mmax, denom,
                  256, 8, 32, stream);
    elu_f32_to_f16<<<((long)NNODES * 256 + 255) / 256, 256, 0, stream>>>(agg, xin, (long)NNODES * 256);

    // layer 3: K=256, H=1, D=64 (no ELU; head-mean is identity for H=1)
    run_gat_layer(W3, b3, attn3, src, dst, xin, wt, h, agg, score, mmax, denom,
                  256, 1, 64, stream);

    // graph max-pool -> [64]
    max_pool<<<64, 256, 0, stream>>>(agg, out, NNODES, 64);
}
